// CopyDecoder_1236950582018
// MI455X (gfx1250) — matrix operand order repaired and checked by reading, not compiled, not run on hardware
//
#include <hip/hip_runtime.h>
#include <hip/hip_bf16.h>
#include <math.h>

#define B_    128
#define S_    256
#define H_    512
#define V_    50000
#define MAXOOV 12
#define VEXT  (V_ + MAXOOV)

typedef __attribute__((ext_vector_type(16))) __bf16 bf16x16;
typedef __attribute__((ext_vector_type(8)))  float  f32x8;

union FragAB { bf16x16 v; unsigned u[8]; };

// fp32 pair -> packed bf16 (hardware convert)
__device__ __forceinline__ unsigned pk2(float a, float b) {
  union { __bf16 h[2]; unsigned u; } p;
  p.h[0] = (__bf16)a;
  p.h[1] = (__bf16)b;
  return p.u;
}

// branch-free transcendentals on the TRANS units (v_exp_f32 / raw v_rcp_f32)
__device__ __forceinline__ float fast_tanh(float x) {
  float e = __expf(2.f * x);                       // inf for large x, 0 for very neg
  return 1.f - 2.f * __builtin_amdgcn_rcpf(e + 1.f);  // saturates to +-1 correctly
}
__device__ __forceinline__ float fast_sigmoid(float x) {
  return __builtin_amdgcn_rcpf(1.f + __expf(-x));
}

// A-fragment K index for 16-bit 16x32 (ISA 7.12.2): lanes 0-15 hold K 0-7 & 16-23
__device__ __forceinline__ int kkA(int j, int hi) {
  return ((j >= 4) ? 16 : 0) + hi * 8 + (j & 3) * 2;
}
// B-fragment K index: same K order as A; lanes 0-15 hold K 0-7 & 16-23
__device__ __forceinline__ int kkB(int j, int hi) {
  return ((j >= 4) ? 16 : 0) + hi * 8 + (j & 3) * 2;
}

// ---------------------------------------------------------------- preamble ---

__global__ void gather_embed(const float* __restrict__ emb,
                             const int* __restrict__ idx,
                             float* __restrict__ xe) {
  int i = blockIdx.x * blockDim.x + threadIdx.x;
  if (i >= B_ * H_) return;
  int b = i / H_, k = i % H_;
  xe[i] = emb[(size_t)idx[b] * H_ + k];
}

// out[m,n] = sum_k A[m, k] * W[n, koff + k] + bias[n]
__global__ void gemm_rowT(const float* __restrict__ A, int lda,
                          const float* __restrict__ W, int ldw, int koff,
                          const float* __restrict__ bias,
                          float* __restrict__ out, int ldo,
                          int M, int N, int K) {
  int idx = blockIdx.x * blockDim.x + threadIdx.x;
  if (idx >= M * N) return;
  int m = idx / N, n = idx % N;
  const float* a = A + (size_t)m * lda;
  const float* w = W + (size_t)n * ldw + koff;
  float s = bias[n];
  for (int k = 0; k < K; k += 4) {
    float4 av = *(const float4*)(a + k);
    float4 wv = *(const float4*)(w + k);
    s += av.x * wv.x + av.y * wv.y + av.z * wv.z + av.w * wv.w;
  }
  out[(size_t)m * ldo + n] = s;
}

__global__ void gru_elem(const float* __restrict__ gi, const float* __restrict__ gh,
                         const float* __restrict__ prev,
                         float* __restrict__ st_out, float* __restrict__ ybuf) {
  int i = blockIdx.x * blockDim.x + threadIdx.x;
  if (i >= B_ * H_) return;
  int b = i / H_, h = i % H_;
  float ir = gi[b * 3 * H_ + h];
  float iz = gi[b * 3 * H_ + H_ + h];
  float in_ = gi[b * 3 * H_ + 2 * H_ + h];
  float hr = gh[b * 3 * H_ + h];
  float hz = gh[b * 3 * H_ + H_ + h];
  float hn = gh[b * 3 * H_ + 2 * H_ + h];
  float p = prev[i];
  float r = fast_sigmoid(ir + hr);
  float z = fast_sigmoid(iz + hz);
  float n = fast_tanh(in_ + r * hn);
  float st = (1.f - z) * n + z * p;
  st_out[i] = st;
  ybuf[(size_t)b * (2 * H_) + h] = st;   // y[:, :H] = state
}

// --------------------------------------------- fused WMMA score kernel ------
// score[b,s] = sum_n tanh( (enc[b,s,:] . W[n, koff:koff+512]) + add[b?,n] ) * mul[b?,n]
// grid = (B, S/32), block = 128 (4 waves). N in {512, 1024}.
__global__ __launch_bounds__(128)
void fused_score_wmma(const float* __restrict__ enc,
                      const float* __restrict__ W, int ldw, int koff,
                      const float* __restrict__ addv, int add_stride,
                      const float* __restrict__ mulv, int mul_stride,
                      const int* __restrict__ mask_idx,
                      float* __restrict__ out_score, int N) {
  __shared__ unsigned short Asm[32 * 520];   // 32 s-rows x 512 K (bf16), padded
  __shared__ unsigned short Bsm[64 * 40];    // 64 n-rows x 32 K  (bf16), padded
  __shared__ float s_part[32][64];           // deterministic per-row reduction

  const int b   = blockIdx.x;
  const int s0  = blockIdx.y * 32;
  const int tid = threadIdx.x;
  const int lane = tid & 31;
  const int wave = tid >> 5;                 // 0..3
  const int lm = lane & 15;
  const int hi = lane >> 4;
  const int nlane = wave * 16 + lm;          // 0..63: this lane's n within chunk

  // stage A: enc[b, s0..s0+31, 0..511] fp32 -> bf16 (uniform trip count)
#pragma unroll
  for (int it = 0; it < 32; ++it) {
    int i = tid + it * 128;
    int row = i >> 7, c4 = i & 127;
    const float4 f = *(const float4*)(enc + ((size_t)(b * S_ + s0 + row)) * H_ + c4 * 4);
    unsigned* d = (unsigned*)&Asm[row * 520 + c4 * 4];
    d[0] = pk2(f.x, f.y);
    d[1] = pk2(f.z, f.w);
  }

  float acc[16];
#pragma unroll
  for (int i = 0; i < 16; ++i) acc[i] = 0.f;

  for (int n0 = 0; n0 < N; n0 += 64) {
    f32x8 cf0, cf1;
#pragma unroll
    for (int i = 0; i < 8; ++i) { cf0[i] = 0.f; cf1[i] = 0.f; }

    for (int kc = 0; kc < H_; kc += 32) {
      __syncthreads();
      // stage B: W[n0+nr, koff+kc .. +31] fp32 -> bf16 (uniform trip count)
#pragma unroll
      for (int it = 0; it < 4; ++it) {
        int i = tid + it * 128;
        int nr = i >> 3, c4 = i & 7;
        const float4 f = *(const float4*)(W + (size_t)(n0 + nr) * ldw + koff + kc + c4 * 4);
        unsigned* d = (unsigned*)&Bsm[nr * 40 + c4 * 4];
        d[0] = pk2(f.x, f.y);
        d[1] = pk2(f.z, f.w);
      }
      __syncthreads();

      FragAB bfr, a0, a1;
#pragma unroll
      for (int j = 0; j < 8; ++j) {
        bfr.u[j] = *(const unsigned*)&Bsm[nlane * 40 + kkB(j, hi)];
        a0.u[j]  = *(const unsigned*)&Asm[lm * 520 + kc + kkA(j, hi)];
        a1.u[j]  = *(const unsigned*)&Asm[(16 + lm) * 520 + kc + kkA(j, hi)];
      }
      cf0 = __builtin_amdgcn_wmma_f32_16x16x32_bf16(false, a0.v, false, bfr.v,
                                                    (short)0, cf0, false, false);
      cf1 = __builtin_amdgcn_wmma_f32_16x16x32_bf16(false, a1.v, false, bfr.v,
                                                    (short)0, cf1, false, false);
    }

    const int n = n0 + nlane;
    const float av = addv[(size_t)b * add_stride + n];
    const float mv = mulv[(size_t)b * mul_stride + n];
#pragma unroll
    for (int i = 0; i < 8; ++i) {
      acc[i]     += fast_tanh(cf0[i] + av) * mv;   // rows hi*8+i
      acc[8 + i] += fast_tanh(cf1[i] + av) * mv;   // rows 16+hi*8+i
    }
  }

  __syncthreads();
#pragma unroll
  for (int i = 0; i < 8; ++i) {
    s_part[hi * 8 + i][nlane]      = acc[i];
    s_part[16 + hi * 8 + i][nlane] = acc[8 + i];
  }
  __syncthreads();
  if (tid < 32) {
    float v = 0.f;
#pragma unroll 8
    for (int c = 0; c < 64; ++c) v += s_part[tid][c];
    int s = s0 + tid;
    if (mask_idx) v += (mask_idx[b * S_ + s] == 0) ? -1000.f : 0.f;
    out_score[b * S_ + s] = v;
  }
}

// ----------------------------------------------------- Wo GEMM (WMMA) -------
// out[b, N0+n] = y[b,:] . Wo[n,:] + Wo_b[n]; out rows stride VEXT. grid covers N.
__global__ __launch_bounds__(256)
void wo_gemm_wmma(const float* __restrict__ Ybuf,
                  const float* __restrict__ Wo,
                  const float* __restrict__ Wob,
                  float* __restrict__ out, int N) {
  __shared__ unsigned short Ysm[128 * 40];   // 128 b-rows x 32 K
  __shared__ unsigned short Wsm[128 * 40];   // 128 n-rows x 32 K

  const int N0  = blockIdx.x * 128;
  const int tid = threadIdx.x;
  const int lane = tid & 31;
  const int wave = tid >> 5;                 // 0..7
  const int lm = lane & 15;
  const int hi = lane >> 4;
  const int nloc = wave * 16 + lm;           // 0..127

  f32x8 cf[8];
#pragma unroll
  for (int t = 0; t < 8; ++t)
#pragma unroll
    for (int i = 0; i < 8; ++i) cf[t][i] = 0.f;

  for (int kc = 0; kc < 2 * H_; kc += 32) {
    __syncthreads();
#pragma unroll
    for (int it = 0; it < 4; ++it) {                 // stage y tile
      int i = tid + it * 256;
      int r = i >> 3, c4 = i & 7;
      const float4 f = *(const float4*)(Ybuf + (size_t)r * (2 * H_) + kc + c4 * 4);
      unsigned* d = (unsigned*)&Ysm[r * 40 + c4 * 4];
      d[0] = pk2(f.x, f.y); d[1] = pk2(f.z, f.w);
    }
#pragma unroll
    for (int it = 0; it < 4; ++it) {                 // stage Wo tile (guarded)
      int i = tid + it * 256;
      int r = i >> 3, c4 = i & 7;
      int gn = N0 + r;
      float4 f = make_float4(0.f, 0.f, 0.f, 0.f);
      if (gn < N) f = *(const float4*)(Wo + (size_t)gn * (2 * H_) + kc + c4 * 4);
      unsigned* d = (unsigned*)&Wsm[r * 40 + c4 * 4];
      d[0] = pk2(f.x, f.y); d[1] = pk2(f.z, f.w);
    }
    // prefetch next K-chunk of the streamed 205MB weight
    if (kc + 32 < 2 * H_) {
      int gn = N0 + (tid >> 1);
      if (gn < N)
        __builtin_prefetch(Wo + (size_t)gn * (2 * H_) + kc + 32 + (tid & 1) * 16, 0, 1);
    }
    __syncthreads();

    FragAB bfr;
#pragma unroll
    for (int j = 0; j < 8; ++j)
      bfr.u[j] = *(const unsigned*)&Wsm[nloc * 40 + kkB(j, hi)];
#pragma unroll
    for (int t = 0; t < 8; ++t) {
      FragAB afr;
#pragma unroll
      for (int j = 0; j < 8; ++j)
        afr.u[j] = *(const unsigned*)&Ysm[(t * 16 + lm) * 40 + kkA(j, hi)];
      cf[t] = __builtin_amdgcn_wmma_f32_16x16x32_bf16(false, afr.v, false, bfr.v,
                                                      (short)0, cf[t], false, false);
    }
  }

  const int n = N0 + nloc;
  if (n < N) {
    const float bias = Wob[n];
#pragma unroll
    for (int t = 0; t < 8; ++t)
#pragma unroll
      for (int i = 0; i < 8; ++i)
        out[(size_t)(t * 16 + hi * 8 + i) * VEXT + n] = cf[t][i] + bias;
  }
}

// --------------------------------------------------------- epilogues --------

__global__ void softmax_s(float* __restrict__ sc) {
  __shared__ float red[256];
  int b = blockIdx.x, tid = threadIdx.x;
  float x = sc[b * S_ + tid];
  red[tid] = x; __syncthreads();
  for (int o = 128; o > 0; o >>= 1) {
    if (tid < o) red[tid] = fmaxf(red[tid], red[tid + o]);
    __syncthreads();
  }
  float m = red[0]; __syncthreads();
  float e = __expf(x - m);
  red[tid] = e; __syncthreads();
  for (int o = 128; o > 0; o >>= 1) {
    if (tid < o) red[tid] += red[tid + o];
    __syncthreads();
  }
  sc[b * S_ + tid] = e / red[0];
}

__global__ void context_k(const float* __restrict__ aw, const float* __restrict__ enc,
                          float* __restrict__ ybuf) {
  int i = blockIdx.x * blockDim.x + threadIdx.x;
  if (i >= B_ * H_) return;
  int b = i / H_, h = i % H_;
  float s = 0.f;
  for (int t = 0; t < S_; ++t)
    s += aw[b * S_ + t] * enc[((size_t)(b * S_ + t)) * H_ + h];
  ybuf[(size_t)b * (2 * H_) + H_ + h] = s;   // y[:, H:] = context
}

__global__ __launch_bounds__(256)
void softmax_big(float* __restrict__ out,            // rows hold raw score_g
                 const float* __restrict__ score_c,
                 float* __restrict__ prob_c,
                 float* __restrict__ pc_sums) {
  __shared__ float red[256];
  const int b = blockIdx.x, tid = threadIdx.x;
  float* row = out + (size_t)b * VEXT;

  float lmax = -3.4e38f;
  for (int i = tid; i < V_; i += 256) lmax = fmaxf(lmax, row[i]);
  float scv = score_c[b * S_ + tid];
  lmax = fmaxf(lmax, scv);
  red[tid] = lmax; __syncthreads();
  for (int o = 128; o > 0; o >>= 1) {
    if (tid < o) red[tid] = fmaxf(red[tid], red[tid + o]);
    __syncthreads();
  }
  const float M = red[0]; __syncthreads();

  float lsum = 0.f;
  for (int i = tid; i < V_; i += 256) lsum += __expf(row[i] - M);
  lsum += __expf(scv - M);
  red[tid] = lsum; __syncthreads();
  for (int o = 128; o > 0; o >>= 1) {
    if (tid < o) red[tid] += red[tid + o];
    __syncthreads();
  }
  const float inv = 1.f / red[0]; __syncthreads();

  for (int i = tid; i < V_; i += 256) row[i] = __expf(row[i] - M) * inv;
  const float pc = __expf(scv - M) * inv;
  prob_c[b * S_ + tid] = pc;
  red[tid] = pc; __syncthreads();
  for (int o = 128; o > 0; o >>= 1) {
    if (tid < o) red[tid] += red[tid + o];
    __syncthreads();
  }
  if (tid == 0) pc_sums[b] = red[0];
  if (tid < MAXOOV) row[V_ + tid] = 1e-4f;
}

__global__ void scatter_pc(float* __restrict__ out, const float* __restrict__ prob_c,
                           const int* __restrict__ eidx) {
  int b = blockIdx.x, s = threadIdx.x;
  atomicAdd(out + (size_t)b * VEXT + eidx[b * S_ + s], prob_c[b * S_ + s]);
}

__global__ __launch_bounds__(256)
void weighted_k(const float* __restrict__ prob_c, const int* __restrict__ eidx,
                const int* __restrict__ input_idx, const float* __restrict__ pc_sums,
                const float* __restrict__ enc, float* __restrict__ wt_out) {
  __shared__ float f[256];
  __shared__ int cnt;
  __shared__ float tot;
  const int b = blockIdx.x, tid = threadIdx.x;
  if (tid == 0) {
    cnt = 0;
    float t = 0.f;
    for (int i = 0; i < B_; ++i) t += pc_sums[i];   // deterministic global sum
    tot = t;
  }
  __syncthreads();
  int match = (eidx[b * S_ + tid] == input_idx[b]) ? 1 : 0;
  if (match) atomicAdd(&cnt, 1);
  __syncthreads();
  const float invp = 1.f / tot;
  const float scale = (cnt > 1) ? (1.f / (float)cnt) : 1.f;
  f[tid] = match ? prob_c[b * S_ + tid] * invp * scale : 0.f;
  __syncthreads();
  for (int h = tid; h < H_; h += 256) {
    float a = 0.f;
    for (int s = 0; s < S_; ++s)
      a += f[s] * enc[((size_t)(b * S_ + s)) * H_ + h];
    wt_out[b * H_ + h] = a;
  }
}

// ------------------------------------------------------------- launch -------

extern "C" void kernel_launch(void* const* d_in, const int* in_sizes, int n_in,
                              void* d_out, int out_size, void* d_ws, size_t ws_size,
                              hipStream_t stream) {
  const int*   input_idx   = (const int*)  d_in[0];
  const float* encoded     = (const float*)d_in[1];
  const int*   encoded_idx = (const int*)  d_in[2];
  const float* prev_state  = (const float*)d_in[3];
  const float* embed       = (const float*)d_in[5];
  const float* Ws_w        = (const float*)d_in[6];
  const float* Ws_b        = (const float*)d_in[7];
  const float* gru_Wih     = (const float*)d_in[8];
  const float* gru_Whh     = (const float*)d_in[9];
  const float* gru_bih     = (const float*)d_in[10];
  const float* gru_bhh     = (const float*)d_in[11];
  const float* attn_W      = (const float*)d_in[12];
  const float* attn_b      = (const float*)d_in[13];
  const float* attn_v      = (const float*)d_in[14];
  const float* Wo_w        = (const float*)d_in[15];
  const float* Wo_b        = (const float*)d_in[16];
  const float* Wc_w        = (const float*)d_in[17];
  const float* Wc_b        = (const float*)d_in[18];

  float* ws      = (float*)d_ws;
  float* xe      = ws;                  // B*H
  float* prevb   = xe + B_ * H_;        // B*H
  float* gi      = prevb + B_ * H_;     // B*3H
  float* gh      = gi + B_ * 3 * H_;    // B*3H
  float* u       = gh + B_ * 3 * H_;    // B*H
  float* ybuf    = u + B_ * H_;         // B*2H
  float* att_sc  = ybuf + B_ * 2 * H_;  // B*S (scores -> attn_w in place)
  float* score_c = att_sc + B_ * S_;    // B*S
  float* prob_c  = score_c + B_ * S_;   // B*S
  float* pc_sums = prob_c + B_ * S_;    // B

  float* out    = (float*)d_out;                 // [B, VEXT]
  float* st_out = out + (size_t)B_ * VEXT;       // [B, H]  (state output)
  float* wt_out = st_out + B_ * H_;              // [B, H]  (weighted output)

  // preamble
  gather_embed<<<(B_ * H_ + 255) / 256, 256, 0, stream>>>(embed, input_idx, xe);
  gemm_rowT<<<(B_ * H_ + 255) / 256, 256, 0, stream>>>(
      prev_state, H_, Ws_w, H_, 0, Ws_b, prevb, H_, B_, H_, H_);
  gemm_rowT<<<(B_ * 3 * H_ + 255) / 256, 256, 0, stream>>>(
      xe, H_, gru_Wih, 2 * H_, 0, gru_bih, gi, 3 * H_, B_, 3 * H_, H_);
  gemm_rowT<<<(B_ * 3 * H_ + 255) / 256, 256, 0, stream>>>(
      prevb, H_, gru_Whh, H_, 0, gru_bhh, gh, 3 * H_, B_, 3 * H_, H_);
  gru_elem<<<(B_ * H_ + 255) / 256, 256, 0, stream>>>(gi, gh, prevb, st_out, ybuf);
  gemm_rowT<<<(B_ * H_ + 255) / 256, 256, 0, stream>>>(
      st_out, H_, attn_W, 2 * H_, 0, attn_b, u, H_, B_, H_, H_);

  // attention scores: add = u[b,n] (state part + bias), mul = attn_v[n]
  fused_score_wmma<<<dim3(B_, S_ / 32), 128, 0, stream>>>(
      encoded, attn_W, 2 * H_, H_, u, H_, attn_v, 0, nullptr, att_sc, H_);
  softmax_s<<<B_, S_, 0, stream>>>(att_sc);
  context_k<<<(B_ * H_ + 255) / 256, 256, 0, stream>>>(att_sc, encoded, ybuf);

  // copy scores: add = Wc_b[n], mul = y[b,n], mask on encoded_idx==0
  fused_score_wmma<<<dim3(B_, S_ / 32), 128, 0, stream>>>(
      encoded, Wc_w, H_, 0, Wc_b, 0, ybuf, 2 * H_, encoded_idx, score_c, 2 * H_);

  // generation scores straight into d_out rows
  wo_gemm_wmma<<<(V_ + 127) / 128, 256, 0, stream>>>(ybuf, Wo_w, Wo_b, out, V_);

  // joint softmax (in place on out rows), copy-prob scatter, weighted output
  softmax_big<<<B_, 256, 0, stream>>>(out, score_c, prob_c, pc_sums);
  scatter_pc<<<B_, S_, 0, stream>>>(out, prob_c, encoded_idx);
  weighted_k<<<B_, S_, 0, stream>>>(prob_c, encoded_idx, input_idx, pc_sums,
                                    encoded, wt_out);
}
